// RetrievalModel_15006615733996
// MI455X (gfx1250) — matrix operand order repaired and checked by reading, not compiled, not run on hardware
//
#include <hip/hip_runtime.h>

typedef __attribute__((ext_vector_type(16))) _Float16 v16h;
typedef __attribute__((ext_vector_type(8)))  _Float16 v8h;
typedef __attribute__((ext_vector_type(8)))  float    v8f;

// ---------------------------------------------------------------------------
// Weight pack: W [d x h] fp32 row-major  ->  Wt [h x kpad] f16 row-major,
// zero-padded in K so every GEMM K is a multiple of 32.
// ---------------------------------------------------------------------------
__global__ void wt_pack_kernel(const float* __restrict__ W, _Float16* __restrict__ Wt,
                               int d, int h, int kpad) {
  int t = blockIdx.x * blockDim.x + threadIdx.x;
  if (t >= h * kpad) return;
  int r = t / kpad;
  int c = t - r * kpad;
  float v = (c < d) ? W[(size_t)c * h + r] : 0.0f;
  Wt[t] = (_Float16)v;
}

// ---------------------------------------------------------------------------
// Embedding gather + concat -> f16 X [B x kpad], padded columns zeroed.
// ---------------------------------------------------------------------------
__global__ void gather_concat_kernel(const int* __restrict__ f0, const int* __restrict__ f1,
                                     const int* __restrict__ f2,
                                     const float* __restrict__ e0, const float* __restrict__ e1,
                                     const float* __restrict__ e2,
                                     _Float16* __restrict__ X,
                                     int nB, int d0, int d1, int d2, int kpad) {
  int t = blockIdx.x * blockDim.x + threadIdx.x;
  if (t >= nB * kpad) return;
  int b = t / kpad;
  int c = t - b * kpad;
  float v = 0.0f;
  if (c < d0)                 v = e0[(size_t)f0[b] * d0 + c];
  else if (c < d0 + d1)       v = e1[(size_t)f1[b] * d1 + (c - d0)];
  else if (c < d0 + d1 + d2)  v = e2[(size_t)f2[b] * d2 + (c - d0 - d1)];
  X[t] = (_Float16)v;
}

// ---------------------------------------------------------------------------
// Register-blocked WMMA GEMM:
//   C[M x N] = relu?( alpha * (A[M x K] @ Bt[N x K]^T + bias) )
// One wave computes a 32x64 output block = 2 M-tiles x 4 N-tiles of 16x16,
// reusing each A fragment across 4 WMMAs and each B fragment across 2.
// Lane layouts chosen here (contiguous K, same for A and B; wave32), v_wmma_f32_16x16x32_f16:
//   A 16x32 f16 : lane L -> row (L&15); half e -> K = hi*16 + k + e,
//                 hi = L>>4.
//   B 32x16 f16 : lane L -> col (L&15); half e -> K = hi*16 + k + e
//                 (contiguous because B is stored transposed, row-major [N x K]).
//   C/D 16x16 f32: VGPR r -> row hi*8 + r, col (L&15).
// All branches are wave-uniform => EXEC all-ones around WMMA.
// Requires M % 32 == 0, N % 64 == 0, K % 32 == 0.
// ---------------------------------------------------------------------------
__global__ __launch_bounds__(256)
void wmma_gemm_kernel(const _Float16* __restrict__ A,   // [M x K] row-major
                      const _Float16* __restrict__ Bt,  // [N x K] row-major (= B^T)
                      const float* __restrict__ bias,   // [N] or nullptr
                      float* __restrict__ C,            // [M x N]
                      int M, int N, int K, float alpha, int relu) {
  int lane = threadIdx.x & 31;
  int wave = threadIdx.x >> 5;
  int wid  = blockIdx.x * (blockDim.x >> 5) + wave;
  int nTiles = N >> 6;                   // 64-wide N blocks
  int tm = wid / nTiles;
  int tn = wid - tm * nTiles;
  if (tm >= (M >> 5)) return;            // wave-uniform

  int r16 = lane & 15;
  int hi  = lane >> 4;

  const _Float16* arow0 = A  + (size_t)(tm * 32 + r16) * K + hi * 16;
  const _Float16* arow1 = arow0 + (size_t)16 * K;
  const _Float16* bc    = Bt + (size_t)(tn * 64 + r16) * K + hi * 16;

  v8f acc[8] = {};                       // acc[mi*4 + ni]
  for (int k = 0; k < K; k += 32) {
    union { v16h v; v8h h[2]; } a0, a1;
    a0.h[0] = *(const v8h*)(arow0 + k);
    a0.h[1] = *(const v8h*)(arow0 + k + 8);
    a1.h[0] = *(const v8h*)(arow1 + k);
    a1.h[1] = *(const v8h*)(arow1 + k + 8);
    v16h b0 = *(const v16h*)(bc + k);
    v16h b1 = *(const v16h*)(bc + (size_t)16 * K + k);
    v16h b2 = *(const v16h*)(bc + (size_t)32 * K + k);
    v16h b3 = *(const v16h*)(bc + (size_t)48 * K + k);

    acc[0] = __builtin_amdgcn_wmma_f32_16x16x32_f16(false, a0.v, false, b0, (short)0, acc[0], false, false);
    acc[1] = __builtin_amdgcn_wmma_f32_16x16x32_f16(false, a0.v, false, b1, (short)0, acc[1], false, false);
    acc[2] = __builtin_amdgcn_wmma_f32_16x16x32_f16(false, a0.v, false, b2, (short)0, acc[2], false, false);
    acc[3] = __builtin_amdgcn_wmma_f32_16x16x32_f16(false, a0.v, false, b3, (short)0, acc[3], false, false);
    acc[4] = __builtin_amdgcn_wmma_f32_16x16x32_f16(false, a1.v, false, b0, (short)0, acc[4], false, false);
    acc[5] = __builtin_amdgcn_wmma_f32_16x16x32_f16(false, a1.v, false, b1, (short)0, acc[5], false, false);
    acc[6] = __builtin_amdgcn_wmma_f32_16x16x32_f16(false, a1.v, false, b2, (short)0, acc[6], false, false);
    acc[7] = __builtin_amdgcn_wmma_f32_16x16x32_f16(false, a1.v, false, b3, (short)0, acc[7], false, false);
  }

#pragma unroll
  for (int mi = 0; mi < 2; ++mi) {
    int row0 = tm * 32 + mi * 16 + hi * 8;
#pragma unroll
    for (int ni = 0; ni < 4; ++ni) {
      int col = tn * 64 + ni * 16 + r16;
      float bv = bias ? bias[col] : 0.0f;
      v8f a = acc[mi * 4 + ni];
#pragma unroll
      for (int r = 0; r < 8; ++r) {
        float v = (a[r] + bv) * alpha;
        if (relu) v = fmaxf(v, 0.0f);
        C[(size_t)(row0 + r) * N + col] = v;
      }
    }
  }
}

// ---------------------------------------------------------------------------
// Per-column batch stats (training-mode BN, biased variance), fused into
// scale/shift: y = x*scale + shift,  scale = gamma*rsqrt(var+eps),
// shift = beta - scale*mean.   One block per column.
// ---------------------------------------------------------------------------
__global__ void col_stats_kernel(const float* __restrict__ Y,
                                 const float* __restrict__ gamma,
                                 const float* __restrict__ beta,
                                 float* __restrict__ scale, float* __restrict__ shift,
                                 int Mrows, int N) {
  __shared__ float s1[256];
  __shared__ float s2[256];
  int j = blockIdx.x;
  float a = 0.0f, b = 0.0f;
  for (int i = threadIdx.x; i < Mrows; i += blockDim.x) {
    float v = Y[(size_t)i * N + j];
    a += v;
    b += v * v;
  }
  s1[threadIdx.x] = a; s2[threadIdx.x] = b;
  __syncthreads();
  for (int s = blockDim.x >> 1; s > 0; s >>= 1) {
    if (threadIdx.x < s) { s1[threadIdx.x] += s1[threadIdx.x + s]; s2[threadIdx.x] += s2[threadIdx.x + s]; }
    __syncthreads();
  }
  if (threadIdx.x == 0) {
    float mean = s1[0] / (float)Mrows;
    float var  = s2[0] / (float)Mrows - mean * mean;
    float inv  = rsqrtf(var + 1e-5f);
    float sc   = gamma[j] * inv;
    scale[j] = sc;
    shift[j] = beta[j] - sc * mean;
  }
}

// BN apply + cast to f16 (next layer's A matrix)
__global__ void bn_cast_kernel(const float* __restrict__ Y,
                               const float* __restrict__ scale, const float* __restrict__ shift,
                               _Float16* __restrict__ Xn, int Mrows, int N) {
  int t = blockIdx.x * blockDim.x + threadIdx.x;
  if (t >= Mrows * N) return;
  int j = t % N;
  Xn[t] = (_Float16)(Y[t] * scale[j] + shift[j]);
}

// Last layer: BN apply + per-row L2 normalize + cast to f16. N fixed at 128.
__global__ void bn_l2norm_kernel(const float* __restrict__ Y,
                                 const float* __restrict__ scale, const float* __restrict__ shift,
                                 _Float16* __restrict__ E) {
  const int N = 128;
  __shared__ float s[128];
  __shared__ float rn;
  int row = blockIdx.x;
  int j = threadIdx.x;
  float v = Y[(size_t)row * N + j] * scale[j] + shift[j];
  s[j] = v * v;
  __syncthreads();
  for (int st = 64; st > 0; st >>= 1) {
    if (j < st) s[j] += s[j + st];
    __syncthreads();
  }
  if (j == 0) rn = 1.0f / fmaxf(sqrtf(s[0]), 1e-12f);
  __syncthreads();
  E[(size_t)row * N + j] = (_Float16)(v * rn);
}

// ---------------------------------------------------------------------------
extern "C" void kernel_launch(void* const* d_in, const int* in_sizes, int n_in,
                              void* d_out, int out_size, void* d_ws, size_t ws_size,
                              hipStream_t stream) {
  (void)in_sizes; (void)n_in; (void)out_size; (void)ws_size;
  const int B = 4096;

  // ---- inputs (setup_inputs dict order, depth-first) ----
  const int*   u_uid = (const int*)d_in[0];
  const int*   u_age = (const int*)d_in[1];
  const int*   u_reg = (const int*)d_in[2];
  const int*   i_iid = (const int*)d_in[3];
  const int*   i_cat = (const int*)d_in[4];
  const int*   i_brd = (const int*)d_in[5];
  const float* Eu0 = (const float*)d_in[6];   // user_id  [1e6 x 64]
  const float* Eu1 = (const float*)d_in[7];   // age      [100 x 16]
  const float* Eu2 = (const float*)d_in[8];   // region   [1000 x 32]
  const float* Ei0 = (const float*)d_in[9];   // item_id  [5e5 x 64]
  const float* Ei1 = (const float*)d_in[10];  // category [1000 x 32]
  const float* Ei2 = (const float*)d_in[11];  // brand    [1e4 x 16]
  // u_mlp layers at 12 + l*4 (W,b,gamma,beta); i_mlp at 24 + l*4.

  const int HIDN[3] = {512, 256, 128};  // layer output dims
  const int KIN[3]  = {128, 512, 256};  // padded input dims (112 -> 128)
  const int DIN[3]  = {112, 512, 256};  // true input dims

  // ---- workspace bump allocator (256B aligned) ----
  char* p = (char*)d_ws;
  auto alloc = [&](size_t bytes) -> void* {
    void* r = (void*)p;
    p += (bytes + 255) & ~(size_t)255;
    return r;
  };
  _Float16* Wt[2][3];
  for (int twr = 0; twr < 2; ++twr)
    for (int l = 0; l < 3; ++l)
      Wt[twr][l] = (_Float16*)alloc((size_t)HIDN[l] * KIN[l] * sizeof(_Float16));
  _Float16* X0 = (_Float16*)alloc((size_t)B * 128 * sizeof(_Float16));
  _Float16* X1 = (_Float16*)alloc((size_t)B * 512 * sizeof(_Float16));
  _Float16* X2 = (_Float16*)alloc((size_t)B * 256 * sizeof(_Float16));
  float* Y     = (float*)alloc((size_t)B * 512 * sizeof(float));
  float* scale = (float*)alloc(512 * sizeof(float));
  float* shift = (float*)alloc(512 * sizeof(float));
  _Float16* Eh[2];
  Eh[0] = (_Float16*)alloc((size_t)B * 128 * sizeof(_Float16));
  Eh[1] = (_Float16*)alloc((size_t)B * 128 * sizeof(_Float16));

  // ---- per-tower pipeline ----
  for (int twr = 0; twr < 2; ++twr) {
    int pbase = 12 + twr * 12;

    // pack weights (transpose + pad, fp32 -> f16)
    for (int l = 0; l < 3; ++l) {
      const float* W = (const float*)d_in[pbase + l * 4 + 0];
      int tot = HIDN[l] * KIN[l];
      wt_pack_kernel<<<(tot + 255) / 256, 256, 0, stream>>>(W, Wt[twr][l], DIN[l], HIDN[l], KIN[l]);
    }

    // embedding gather + concat -> X0 [B x 128] f16 (cols 112..127 = 0)
    if (twr == 0)
      gather_concat_kernel<<<(B * 128 + 255) / 256, 256, 0, stream>>>(
          u_uid, u_age, u_reg, Eu0, Eu1, Eu2, X0, B, 64, 16, 32, 128);
    else
      gather_concat_kernel<<<(B * 128 + 255) / 256, 256, 0, stream>>>(
          i_iid, i_cat, i_brd, Ei0, Ei1, Ei2, X0, B, 64, 32, 16, 128);

    _Float16* Xin[3] = {X0, X1, X2};
    for (int l = 0; l < 3; ++l) {
      const float* bb = (const float*)d_in[pbase + l * 4 + 1];
      const float* gg = (const float*)d_in[pbase + l * 4 + 2];
      const float* be = (const float*)d_in[pbase + l * 4 + 3];
      int N = HIDN[l], K = KIN[l];

      // waves = (M/32) * (N/64); divisible by 8 for all sizes here
      int waves = (B / 32) * (N / 64);
      wmma_gemm_kernel<<<waves / 8, 256, 0, stream>>>(
          Xin[l], Wt[twr][l], bb, Y, B, N, K, 1.0f, /*relu=*/1);

      col_stats_kernel<<<N, 256, 0, stream>>>(Y, gg, be, scale, shift, B, N);

      if (l < 2)
        bn_cast_kernel<<<(B * N + 255) / 256, 256, 0, stream>>>(Y, scale, shift, Xin[l + 1], B, N);
      else
        bn_l2norm_kernel<<<B, 128, 0, stream>>>(Y, scale, shift, Eh[twr]);
    }
  }

  // ---- logits = (ue @ ie^T) / TEMP : [4096 x 4096] fp32 ----
  {
    int waves = (B / 32) * (B / 64);          // 8192 waves -> 1024 blocks
    wmma_gemm_kernel<<<waves / 8, 256, 0, stream>>>(
        Eh[0], Eh[1], nullptr, (float*)d_out, B, B, 128, 10.0f, /*relu=*/0);
  }
}
